// TransformerBlock_9345848836486
// MI455X (gfx1250) — matrix operand order repaired and checked by reading, not compiled, not run on hardware
//
#include <hip/hip_runtime.h>
#include <math.h>

// ---------------- types / constants ----------------
typedef __bf16 bf16;
typedef __attribute__((ext_vector_type(16))) __bf16 v16bf;
typedef __attribute__((ext_vector_type(8)))  __bf16 v8bf;
typedef __attribute__((ext_vector_type(8)))  float  v8f;

static constexpr int Bd = 4, Td = 128, Od = 4, Nd = 32, Dd = 256, Md = 1024;
static constexpr int NHd = 8;                 // heads, head_dim = 32
static constexpr int TOK = Bd * Td * Od * Nd; // 65536 tokens

// ---------------- scalar helpers ----------------
__device__ __forceinline__ bf16 f2bf(float f) {
  union { float f; unsigned u; } a; a.f = f;
  unsigned r = (a.u + 0x7fffu + ((a.u >> 16) & 1u)) >> 16;   // RNE
  union { unsigned short s; bf16 b; } o; o.s = (unsigned short)r;
  return o.b;
}

__device__ __forceinline__ v16bf cat8(v8bf lo, v8bf hi) {
  return __builtin_shufflevector(lo, hi, 0,1,2,3,4,5,6,7,8,9,10,11,12,13,14,15);
}

// A fragment: 16(M) x 32(K) bf16 tile from row-major [*, ld] at (row0=0, k0=0) of p.
// ISA 7.12.2: lanes 0-15 -> elems 0..7 = K 0..7, 8..15 = K 16..23 (M = lane);
//             lanes 16-31 -> K 8..15 / 24..31.
__device__ __forceinline__ v16bf load_a(const bf16* p, int ld) {
  int lane = threadIdx.x & 31;
  const bf16* q = p + (lane & 15) * ld + ((lane < 16) ? 0 : 8);
  v8bf lo = *(const v8bf*)(q);
  v8bf hi = *(const v8bf*)(q + 16);
  return cat8(lo, hi);
}

// B fragment: 32(K) x 16(N). Element (k,n) at wt[n*ldw + k] (i.e. W^T access of
// a row-major weight W[N,K]). lanes 0-15: K 0..7 / 16..23; lanes 16-31: K 8..15 / 24..31; n = lane%16.
__device__ __forceinline__ v16bf load_b(const bf16* wt, int ldw) {
  int lane = threadIdx.x & 31;
  const bf16* q = wt + (lane & 15) * ldw + ((lane < 16) ? 0 : 8);
  v8bf lo = *(const v8bf*)(q);
  v8bf hi = *(const v8bf*)(q + 16);
  return cat8(lo, hi);
}

__device__ __forceinline__ v8f wmma_bf16(v16bf a, v16bf b, v8f c) {
  return __builtin_amdgcn_wmma_f32_16x16x32_bf16(false, a, false, b, (short)0, c,
                                                 false, false);
}

__device__ __forceinline__ float grp_max16(float v) {
  #pragma unroll
  for (int m = 1; m < 16; m <<= 1) v = fmaxf(v, __shfl_xor(v, m, 32));
  return v;
}
__device__ __forceinline__ float grp_sum16(float v) {
  #pragma unroll
  for (int m = 1; m < 16; m <<= 1) v += __shfl_xor(v, m, 32);
  return v;
}

// ---------------- kernels ----------------

// fp32 -> bf16 convert (weights)
__global__ void k_cvt(const float* __restrict__ s, bf16* __restrict__ d, int n) {
  int i = blockIdx.x * blockDim.x + threadIdx.x;
  if (i < n) d[i] = f2bf(s[i]);
}

// LayerNorm, wave per token. PERM=0: identity rows. PERM=1: output row is the
// temporal-permuted row ((b*O+o)*N+n)*T + t, gathered from the original layout.
template <int PERM>
__global__ __launch_bounds__(256) void k_ln(const float* __restrict__ x,
                                            const float* __restrict__ g,
                                            const float* __restrict__ bb,
                                            bf16* __restrict__ out) {
  int wid  = (int)((blockIdx.x * blockDim.x + threadIdx.x) >> 5);
  int lane = threadIdx.x & 31;
  int r_in;
  if (PERM == 0) {
    r_in = wid;
  } else {
    int t = wid % Td, bon = wid / Td;
    int n = bon % Nd, o = (bon / Nd) % Od, b = bon / (Nd * Od);
    r_in = ((b * Td + t) * Od + o) * Nd + n;
  }
  const float* xr = x + (size_t)r_in * Dd;
  float v[8];
  float s = 0.f;
  #pragma unroll
  for (int i = 0; i < 8; i++) { v[i] = xr[i * 32 + lane]; s += v[i]; }
  #pragma unroll
  for (int m = 1; m < 32; m <<= 1) s += __shfl_xor(s, m, 32);
  float mean = s * (1.0f / Dd);
  float var = 0.f;
  #pragma unroll
  for (int i = 0; i < 8; i++) { float d0 = v[i] - mean; var += d0 * d0; }
  #pragma unroll
  for (int m = 1; m < 32; m <<= 1) var += __shfl_xor(var, m, 32);
  float rstd = rsqrtf(var * (1.0f / Dd) + 1e-5f);
  bf16* orow = out + (size_t)wid * Dd;
  #pragma unroll
  for (int i = 0; i < 8; i++) {
    int c = i * 32 + lane;
    orow[c] = f2bf((v[i] - mean) * rstd * g[c] + bb[c]);
  }
}

// QKV projection: A[TOK,256] bf16 @ {Wq,Wk,Wv}^T + bias.
// One wave per 16x64 output tile (A fragment reused across 4 B fragments).
// Scatters Q,K into [BH,S,32] and V transposed into [BH,32,S].
template <int S>
__global__ __launch_bounds__(256) void k_qkv(
    const bf16* __restrict__ A, const bf16* __restrict__ wq,
    const bf16* __restrict__ wk, const bf16* __restrict__ wv,
    const float* __restrict__ bq, const float* __restrict__ bk,
    const float* __restrict__ bv, bf16* __restrict__ qb,
    bf16* __restrict__ kb2, bf16* __restrict__ vtb) {
  int gw   = (int)((blockIdx.x * blockDim.x + threadIdx.x) >> 5);
  int lane = threadIdx.x & 31;
  const int CG = 768 / 64;        // 12 column groups of 64
  int mt = gw / CG, cg = gw % CG;
  int which = cg >> 2;            // 0=Q 1=K 2=V  (4 groups of 64 per matrix)
  int c0 = (cg & 3) * 64;         // column within the 256-wide matrix
  const bf16* W = (which == 0) ? wq : (which == 1) ? wk : wv;
  const float* bias = (which == 0) ? bq : (which == 1) ? bk : bv;
  const bf16* Abase = A + (size_t)(mt * 16) * Dd;
  v8f acc[4] = {};
  #pragma unroll
  for (int kk = 0; kk < Dd / 32; kk++) {
    v16bf a = load_a(Abase + kk * 32, Dd);
    #pragma unroll
    for (int nt = 0; nt < 4; nt++) {
      v16bf b = load_b(W + (size_t)(c0 + nt * 16) * Dd + kk * 32, Dd);
      acc[nt] = wmma_bf16(a, b, acc[nt]);
    }
  }
  int rbase = mt * 16 + ((lane >> 4) << 3);
  #pragma unroll
  for (int nt = 0; nt < 4; nt++) {
    int n = c0 + nt * 16 + (lane & 15);
    float bn = bias[n];
    int head = n >> 5, hd = n & 31;
    #pragma unroll
    for (int v = 0; v < 8; v++) {
      int r = rbase + v;
      int bseq = r / S, t = r % S;
      int bh = bseq * NHd + head;
      bf16 val = f2bf(acc[nt][v] + bn);
      if (which == 2)
        vtb[((size_t)bh * 32 + hd) * S + t] = val;        // V^T [BH,32,S]
      else if (which == 0)
        qb[((size_t)bh * S + t) * 32 + hd] = val;         // Q   [BH,S,32]
      else
        kb2[((size_t)bh * S + t) * 32 + hd] = val;        // K   [BH,S,32]
    }
  }
}

// Attention: one wave owns a 16-row Q strip; head_dim 32 == WMMA K.
// Scores kept in registers (S/16 accumulators), softmax via 16-lane shuffles,
// P restaged through LDS (C-layout -> A-layout), P@V via WMMA on V^T.
template <int S>
__global__ __launch_bounds__(256) void k_attn(const bf16* __restrict__ q,
                                              const bf16* __restrict__ k,
                                              const bf16* __restrict__ vt,
                                              bf16* __restrict__ out) {
  constexpr int STRIPS = S / 16;
  constexpr int JT = S / 16;
  __shared__ bf16 lp[8 * 16 * S];
  int widx = threadIdx.x >> 5, lane = threadIdx.x & 31;
  int gw = blockIdx.x * 8 + widx;
  int bh = gw / STRIPS;
  int q0 = (gw % STRIPS) * 16;

  v16bf af = load_a(q + ((size_t)bh * S + q0) * 32, 32);
  v8f sc[JT];
  #pragma unroll
  for (int j = 0; j < JT; j++) {
    v16bf bfrag = load_b(k + ((size_t)bh * S + j * 16) * 32, 32);
    v8f z = {};
    sc[j] = wmma_bf16(af, bfrag, z);
  }
  const float scale = 0.17677669529663687f;  // 1/sqrt(32)
  bf16* myp = lp + widx * 16 * S;
  int row0 = (lane >> 4) << 3;
  #pragma unroll
  for (int v = 0; v < 8; v++) {
    float m = -1e30f;
    #pragma unroll
    for (int j = 0; j < JT; j++) m = fmaxf(m, sc[j][v]);
    m = grp_max16(m);
    float s = 0.f;
    #pragma unroll
    for (int j = 0; j < JT; j++) {
      float e = __expf(scale * (sc[j][v] - m));
      sc[j][v] = e;
      s += e;
    }
    s = grp_sum16(s);
    float rs = 1.0f / s;
    int row = row0 + v;
    #pragma unroll
    for (int j = 0; j < JT; j++)
      myp[row * S + j * 16 + (lane & 15)] = f2bf(sc[j][v] * rs);
  }
  // P @ V   (per-wave LDS region; DS ops are in-order within a wave)
  const bf16* vbase = vt + (size_t)bh * 32 * S;
  v8f o0 = {}, o1 = {};
  #pragma unroll
  for (int kk = 0; kk < S / 32; kk++) {
    v16bf pa = load_a(myp + kk * 32, S);
    v16bf b0 = load_b(vbase + kk * 32, S);
    v16bf b1 = load_b(vbase + (size_t)16 * S + kk * 32, S);
    o0 = wmma_bf16(pa, b0, o0);
    o1 = wmma_bf16(pa, b1, o1);
  }
  int bseq = bh / NHd, head = bh % NHd;
  int col0 = head * 32 + (lane & 15);
  #pragma unroll
  for (int v = 0; v < 8; v++) {
    int t = q0 + row0 + v;
    size_t ro = ((size_t)bseq * S + t) * Dd;
    out[ro + col0]      = f2bf(o0[v]);
    out[ro + col0 + 16] = f2bf(o1[v]);
  }
}

// Output projection + residual, one wave per 16x64 tile. PERM=1: un-permute
// temporal rows back to the original token layout; PERM=0: identity.
template <int PERM>
__global__ __launch_bounds__(256) void k_oproj(const bf16* __restrict__ A,
                                               const bf16* __restrict__ W,
                                               const float* __restrict__ bias,
                                               const float* __restrict__ resid,
                                               float* __restrict__ out) {
  int gw = (int)((blockIdx.x * blockDim.x + threadIdx.x) >> 5);
  int lane = threadIdx.x & 31;
  const int CG = Dd / 64;         // 4 column groups
  int mt = gw / CG, cg = gw % CG;
  int c0 = cg * 64;
  const bf16* Abase = A + (size_t)(mt * 16) * Dd;
  v8f acc[4] = {};
  #pragma unroll
  for (int kk = 0; kk < Dd / 32; kk++) {
    v16bf a = load_a(Abase + kk * 32, Dd);
    #pragma unroll
    for (int nt = 0; nt < 4; nt++) {
      v16bf b = load_b(W + (size_t)(c0 + nt * 16) * Dd + kk * 32, Dd);
      acc[nt] = wmma_bf16(a, b, acc[nt]);
    }
  }
  int rbase = mt * 16 + ((lane >> 4) << 3);
  #pragma unroll
  for (int v = 0; v < 8; v++) {
    int r = rbase + v, orig;
    if (PERM == 0) {
      orig = r;
    } else {
      int t = r % Td, bon = r / Td;
      int n2 = bon % Nd, o = (bon / Nd) % Od, b = bon / (Nd * Od);
      orig = ((b * Td + t) * Od + o) * Nd + n2;
    }
    size_t ro = (size_t)orig * Dd;
    #pragma unroll
    for (int nt = 0; nt < 4; nt++) {
      int n = c0 + nt * 16 + (lane & 15);
      out[ro + n] = resid[ro + n] + acc[nt][v] + bias[n];
    }
  }
}

// Fused MLP: 16 rows/block; hidden h (16x1024 bf16) never leaves LDS.
__global__ __launch_bounds__(256) void k_mlp(const float* __restrict__ x,
                                             const bf16* __restrict__ w1,
                                             const float* __restrict__ b1,
                                             const bf16* __restrict__ w2,
                                             const float* __restrict__ b2,
                                             float* __restrict__ out) {
  __shared__ bf16 xa[16 * Dd];   // 8 KB
  __shared__ bf16 h[16 * Md];    // 32 KB
  int tid = threadIdx.x, widx = tid >> 5, lane = tid & 31;
  size_t r0 = (size_t)blockIdx.x * 16;
  // phase 0: load 16x256 fp32 rows, convert to bf16 in LDS
  const float4* xs = (const float4*)(x + r0 * Dd);
  #pragma unroll
  for (int i = 0; i < 4; i++) {
    int e = i * 256 + tid;        // 1024 float4 total
    float4 f = xs[e];
    int base = e * 4;
    xa[base + 0] = f2bf(f.x); xa[base + 1] = f2bf(f.y);
    xa[base + 2] = f2bf(f.z); xa[base + 3] = f2bf(f.w);
  }
  __syncthreads();
  int row0 = (lane >> 4) << 3;
  // phase 1: h = gelu_exact(x @ w1^T + b1); wave owns 128 hidden cols.
  // A fragments hoisted: one LDS read of the 16x256 strip, reused for all 8 tiles.
  v16bf afr[8];
  #pragma unroll
  for (int kk = 0; kk < Dd / 32; kk++) afr[kk] = load_a(xa + kk * 32, Dd);
  #pragma unroll
  for (int nt = 0; nt < 8; nt++) {
    int c0 = widx * 128 + nt * 16;
    v8f acc = {};
    #pragma unroll
    for (int kk = 0; kk < Dd / 32; kk++) {
      v16bf b = load_b(w1 + (size_t)c0 * Dd + kk * 32, Dd);
      acc = wmma_bf16(afr[kk], b, acc);
    }
    int n = c0 + (lane & 15);
    float bn = b1[n];
    #pragma unroll
    for (int v = 0; v < 8; v++) {
      float t = acc[v] + bn;
      float g = 0.5f * t * (1.0f + erff(t * 0.70710678118654752f));
      h[(row0 + v) * Md + n] = f2bf(g);
    }
  }
  __syncthreads();
  // phase 2: out = x + h @ w2^T + b2; wave owns 32 output cols
  #pragma unroll
  for (int nt = 0; nt < 2; nt++) {
    int c0 = widx * 32 + nt * 16;
    v8f acc = {};
    #pragma unroll
    for (int kk = 0; kk < Md / 32; kk++) {
      v16bf a = load_a(h + kk * 32, Md);
      v16bf b = load_b(w2 + (size_t)c0 * Md + kk * 32, Md);
      acc = wmma_bf16(a, b, acc);
    }
    int n = c0 + (lane & 15);
    float bn = b2[n];
    #pragma unroll
    for (int v = 0; v < 8; v++) {
      size_t idx = (r0 + row0 + v) * Dd + n;
      out[idx] = x[idx] + acc[v] + bn;
    }
  }
}

// ---------------- host launcher ----------------
extern "C" void kernel_launch(void* const* d_in, const int* in_sizes, int n_in,
                              void* d_out, int out_size, void* d_ws, size_t ws_size,
                              hipStream_t stream) {
  (void)in_sizes; (void)n_in; (void)out_size; (void)ws_size;
  const float* x   = (const float*)d_in[0];
  const float* tqw = (const float*)d_in[1];  const float* tqb = (const float*)d_in[2];
  const float* tkw = (const float*)d_in[3];  const float* tkb = (const float*)d_in[4];
  const float* tvw = (const float*)d_in[5];  const float* tvb = (const float*)d_in[6];
  const float* tow = (const float*)d_in[7];  const float* tob = (const float*)d_in[8];
  const float* tng = (const float*)d_in[9];  const float* tnb = (const float*)d_in[10];
  const float* sqw = (const float*)d_in[11]; const float* sqb = (const float*)d_in[12];
  const float* skw = (const float*)d_in[13]; const float* skb = (const float*)d_in[14];
  const float* svw = (const float*)d_in[15]; const float* svb = (const float*)d_in[16];
  const float* sow = (const float*)d_in[17]; const float* sob = (const float*)d_in[18];
  const float* sng = (const float*)d_in[19]; const float* snb = (const float*)d_in[20];
  const float* f1w = (const float*)d_in[21]; const float* f1b = (const float*)d_in[22];
  const float* f2w = (const float*)d_in[23]; const float* f2b = (const float*)d_in[24];

  const size_t DDn = (size_t)Dd * Dd;      // 65536
  const size_t MDn = (size_t)Md * Dd;      // 262144
  const size_t QEn = (size_t)TOK * Dd;     // 16,777,216 (= BH*S*32 per stage)

  char* p = (char*)d_ws;
  auto take = [&](size_t bytes) -> char* {
    char* r = p; p += (bytes + 255) & ~(size_t)255; return r;
  };
  bf16* w_tq = (bf16*)take(DDn * 2); bf16* w_tk = (bf16*)take(DDn * 2);
  bf16* w_tv = (bf16*)take(DDn * 2); bf16* w_to = (bf16*)take(DDn * 2);
  bf16* w_sq = (bf16*)take(DDn * 2); bf16* w_sk = (bf16*)take(DDn * 2);
  bf16* w_sv = (bf16*)take(DDn * 2); bf16* w_so = (bf16*)take(DDn * 2);
  bf16* w_f1 = (bf16*)take(MDn * 2); bf16* w_f2 = (bf16*)take(MDn * 2);
  bf16* lnb  = (bf16*)take(QEn * 2);
  bf16* qb   = (bf16*)take(QEn * 2);
  bf16* kb   = (bf16*)take(QEn * 2);
  bf16* vtb  = (bf16*)take(QEn * 2);
  bf16* atb  = (bf16*)take(QEn * 2);
  float* x1  = (float*)take(QEn * 4);
  float* x2  = (float*)take(QEn * 4);

  auto cvt = [&](const float* s, bf16* d, size_t n) {
    k_cvt<<<(int)((n + 255) / 256), 256, 0, stream>>>(s, d, (int)n);
  };
  cvt(tqw, w_tq, DDn); cvt(tkw, w_tk, DDn); cvt(tvw, w_tv, DDn); cvt(tow, w_to, DDn);
  cvt(sqw, w_sq, DDn); cvt(skw, w_sk, DDn); cvt(svw, w_sv, DDn); cvt(sow, w_so, DDn);
  cvt(f1w, w_f1, MDn); cvt(f2w, w_f2, MDn);

  const int LN_BLK  = TOK / 8;                        // 8192 (wave per token)
  const int QKV_BLK = (TOK / 16) * (768 / 64) / 8;    // 6144 (wave per 16x64 tile)
  const int OP_BLK  = (TOK / 16) * (Dd / 64) / 8;     // 2048 (wave per 16x64 tile)
  const int ATT_BLK = 4096;                           // both stages: 32768 waves

  // ---- temporal attention (sequences over T, batch B*O*N) ----
  k_ln<1><<<LN_BLK, 256, 0, stream>>>(x, tng, tnb, lnb);
  k_qkv<Td><<<QKV_BLK, 256, 0, stream>>>(lnb, w_tq, w_tk, w_tv, tqb, tkb, tvb,
                                         qb, kb, vtb);
  k_attn<Td><<<ATT_BLK, 256, 0, stream>>>(qb, kb, vtb, atb);
  k_oproj<1><<<OP_BLK, 256, 0, stream>>>(atb, w_to, tob, x, x1);

  // ---- spatial attention (sequences over N, batch B*T*O; identity layout) ----
  k_ln<0><<<LN_BLK, 256, 0, stream>>>(x1, sng, snb, lnb);
  k_qkv<Nd><<<QKV_BLK, 256, 0, stream>>>(lnb, w_sq, w_sk, w_sv, sqb, skb, svb,
                                         qb, kb, vtb);
  k_attn<Nd><<<ATT_BLK, 256, 0, stream>>>(qb, kb, vtb, atb);
  k_oproj<0><<<OP_BLK, 256, 0, stream>>>(atb, w_so, sob, x1, x2);

  // ---- fused MLP with residual ----
  k_mlp<<<TOK / 16, 256, 0, stream>>>(x2, w_f1, f1b, w_f2, f2b, (float*)d_out);
}
